// DeformableTransformer_61641370632777
// MI455X (gfx1250) — compile-verified, matrix operand order checked by reading, not run on hardware
//
#include <hip/hip_runtime.h>
#include <math.h>
#include <stdint.h>

// Problem constants (match reference)
#define B_      8
#define Q_      300
#define C_      256
#define NH_     8
#define DH_     32
#define NL_     4
#define NP_     4
#define DFF_    1024
#define NLAYERS_ 6
#define LV_     21760
#define BQ_     (B_*Q_)

typedef __attribute__((ext_vector_type(16))) _Float16 v16h;
typedef __attribute__((ext_vector_type(8)))  _Float16 v8h;
typedef __attribute__((ext_vector_type(8)))  float    v8f;
typedef __attribute__((ext_vector_type(4)))  float    v4f;
typedef __attribute__((ext_vector_type(4)))  unsigned u32x4;
typedef __attribute__((ext_vector_type(8)))  unsigned u32x8;

__device__ __forceinline__ int imin(int a, int b) { return a < b ? a : b; }
__device__ __forceinline__ int imax(int a, int b) { return a > b ? a : b; }

// ---------------------------------------------------------------------------
// Tensor Data Mover: DMA a 32-row x 64-col f32 tile of W (row stride N elems)
// from global memory into LDS at lds_addr, padding +1 dword every 64 dwords
// so the LDS row stride is 65 dwords (bank-conflict-free column reads).
// 2-D tile -> 2 descriptor groups (VADDR2/VADDR3 NULL form).
// ---------------------------------------------------------------------------
__device__ __forceinline__ void tdm_load_w_tile(const float* gsrc,
                                                unsigned lds_addr, int N)
{
    unsigned long long ga = (unsigned long long)(uintptr_t)gsrc;
    u32x4 g0;
    g0[0] = 1u;                                   // count=1 valid user D#
    g0[1] = lds_addr;                             // LDS dest (bytes)
    g0[2] = (unsigned)ga;                         // global_addr[31:0]
    g0[3] = ((unsigned)(ga >> 32) & 0x01FFFFFFu)  // global_addr[56:32]
            | (2u << 30);                         // type=2 ("image")
    u32x8 g1;
    g1[0] = (2u << 16)                            // data_size = 4 bytes
          | (1u << 20)                            // pad_enable
          | (5u << 22);                           // pad_interval: 64 dwords
                                                  // pad_amount code 0: 1 dword
    g1[1] = ((unsigned)N & 0xFFFFu) << 16;        // tensor_dim0[15:0]
    g1[2] = ((unsigned)N >> 16) | (32u << 16);    // tensor_dim0[31:16] | tensor_dim1[15:0]=32
    g1[3] = (64u << 16);                          // tensor_dim1[31:16]=0 | tile_dim0=64
    g1[4] = 32u;                                  // tile_dim1=32, tile_dim2=0 (2-D)
    g1[5] = (unsigned)N;                          // tensor_dim0_stride[31:0] (elems)
    g1[6] = 0u;                                   // stride[47:32] | dim1_stride lo
    g1[7] = 0u;
    asm volatile("tensor_load_to_lds %0, %1" :: "s"(g0), "s"(g1) : "memory");
}

// ---------------------------------------------------------------------------
// WMMA GEMM:  Y[M,N] = relu?( X[M,K] @ W[K,N] + bias[N] )
// Block = 128 threads = 4 waves; each wave owns a 16x64 output strip
// (one A fragment reused across 4 WMMAs). W tiles are DMA'd by the TDM into
// a double-buffered LDS tile, overlapping DMA of tile k+1 with math on tile k.
// Requires: K % 32 == 0, N % 64 == 0, M % 16 == 0 (true for all call sites).
// ---------------------------------------------------------------------------
#define WROW 65   // LDS dwords per staged W row (64 data + 1 TDM pad)

__global__ void __launch_bounds__(128)
wmma_gemm_bias(const float* __restrict__ X, const float* __restrict__ Wm,
               const float* __restrict__ bias, float* __restrict__ Y,
               int M, int N, int K, int relu)
{
    __shared__ __align__(16) float sWf[2][32 * WROW];

    const int tid   = threadIdx.x;
    const int wave  = tid >> 5;
    const int lane  = tid & 31;
    const int half  = lane >> 4;   // which 16-lane half of the wave
    const int mrow  = lane & 15;   // row within A tile / col within B,C,D tile
    const int tileM = blockIdx.x * 4 + wave;
    const int tileN = blockIdx.y;          // covers 64 output columns

    // clamped A row so out-of-range waves still issue valid loads
    const int arow = imin(tileM * 16 + mrow, M - 1);
    const float* xrow  = X + (size_t)arow * K;
    const float* wtile = Wm + tileN * 64;  // column offset of this tile strip

    const unsigned lds0 = (unsigned)(uintptr_t)&sWf[0][0];
    const unsigned lds1 = (unsigned)(uintptr_t)&sWf[1][0];

    // prologue: DMA first W tile (one wave issues; descriptors are uniform)
    if (wave == 0)
        tdm_load_w_tile(wtile, lds0, N);

    v8f acc[4] = {};

    for (int k0 = 0; k0 < K; k0 += 32) {
        const int cur = (k0 >> 5) & 1;
        if (wave == 0) {
            if (k0 + 32 < K) {
                // DMA next tile into the other buffer, then retire current
                tdm_load_w_tile(wtile + (size_t)(k0 + 32) * N,
                                cur ? lds0 : lds1, N);
                __builtin_amdgcn_s_wait_tensorcnt(1);
            } else {
                __builtin_amdgcn_s_wait_tensorcnt(0);
            }
        }
        __syncthreads();   // current tile visible to all 4 waves

        // A fragment: per-lane row arow; K layout per ISA:
        //   elems 0..7  -> K = half*8 + 0..7
        //   elems 8..15 -> K = 16 + half*8 + 0..7
        const v4f* pa0 = (const v4f*)(xrow + k0 + half * 8);
        const v4f* pa1 = (const v4f*)(xrow + k0 + 16 + half * 8);
        v4f f0 = pa0[0], f1 = pa0[1], f2 = pa1[0], f3 = pa1[1];
        v16h a;
        #pragma unroll
        for (int j = 0; j < 4; ++j) {
            a[j]      = (_Float16)f0[j];
            a[4 + j]  = (_Float16)f1[j];
            a[8 + j]  = (_Float16)f2[j];
            a[12 + j] = (_Float16)f3[j];
        }

        // 4 B fragments from the padded row-major f32 tile (cvt at use)
        const float* bufc = &sWf[cur][0];
        #pragma unroll
        for (int t = 0; t < 4; ++t) {
            const int nc = t * 16 + mrow;
            v16h b;
            #pragma unroll
            for (int j = 0; j < 16; ++j) {
                const int kk = ((j & 8) << 1) + half * 8 + (j & 7);
                b[j] = (_Float16)bufc[kk * WROW + nc];
            }
            acc[t] = __builtin_amdgcn_wmma_f32_16x16x32_f16(
                         false, a, false, b, (short)0, acc[t], false, false);
        }
        __syncthreads();   // done with buf[cur^1]'s successor overwrite target
    }

    // Epilogue: C/D layout -> element r of acc is row (r + 8*half), col mrow
    #pragma unroll
    for (int t = 0; t < 4; ++t) {
        int n = tileN * 64 + t * 16 + mrow;
        float bv = bias[n];
        #pragma unroll
        for (int r = 0; r < 8; ++r) {
            int m = tileM * 16 + r + half * 8;
            if (m < M) {
                float o = acc[t][r] + bv;
                if (relu) o = fmaxf(o, 0.f);
                Y[(size_t)m * N + n] = o;
            }
        }
    }
}

// ---------------------------------------------------------------------------
// out = a + b (elementwise)
// ---------------------------------------------------------------------------
__global__ void add_kernel(float* __restrict__ out, const float* __restrict__ a,
                           const float* __restrict__ b, int n)
{
    int i = blockIdx.x * blockDim.x + threadIdx.x;
    if (i < n) out[i] = a[i] + b[i];
}

// ---------------------------------------------------------------------------
// y = LayerNorm(a + b) * g + beta, one 256-wide row per 256-thread block
// ---------------------------------------------------------------------------
__global__ void __launch_bounds__(256)
res_ln_kernel(const float* __restrict__ A, const float* __restrict__ Bv,
              const float* __restrict__ g, const float* __restrict__ be,
              float* __restrict__ Y)
{
    __shared__ float red[256];
    const int row = blockIdx.x, t = threadIdx.x;
    float v = A[(size_t)row * C_ + t] + Bv[(size_t)row * C_ + t];

    red[t] = v; __syncthreads();
    #pragma unroll
    for (int s = 128; s > 0; s >>= 1) { if (t < s) red[t] += red[t + s]; __syncthreads(); }
    float mean = red[0] * (1.f / C_);
    __syncthreads();

    float d = v - mean;
    red[t] = d * d; __syncthreads();
    #pragma unroll
    for (int s = 128; s > 0; s >>= 1) { if (t < s) red[t] += red[t + s]; __syncthreads(); }
    float var = red[0] * (1.f / C_);

    Y[(size_t)row * C_ + t] = d * rsqrtf(var + 1e-5f) * g[t] + be[t];
}

// ---------------------------------------------------------------------------
// Self-attention core: one wave per (b,h,q), lane = channel d (DH==32==wave32)
// Online (flash-style) softmax; scores via cross-lane tree reduction.
// Layouts: Q/K/V are (B*Q, NH*DH) row-major.
// ---------------------------------------------------------------------------
__global__ void __launch_bounds__(256)
self_attn_kernel(const float* __restrict__ Qb, const float* __restrict__ Kb,
                 const float* __restrict__ Vb, float* __restrict__ O)
{
    const int gw   = (blockIdx.x * blockDim.x + threadIdx.x) >> 5;
    const int lane = threadIdx.x & 31;
    if (gw >= B_ * NH_ * Q_) return;
    const int b = gw / (NH_ * Q_);
    const int r = gw % (NH_ * Q_);
    const int h = r / Q_;
    const int q = r % Q_;

    const float scale = 0.17677669529663687f;  // 1/sqrt(32)
    const float qd = Qb[((size_t)(b * Q_ + q)) * C_ + h * DH_ + lane] * scale;

    float m = -INFINITY, l = 0.f, acc = 0.f;
    for (int k = 0; k < Q_; ++k) {
        size_t base = ((size_t)(b * Q_ + k)) * C_ + h * DH_ + lane;
        float s = qd * Kb[base];
        #pragma unroll
        for (int off = 16; off > 0; off >>= 1) s += __shfl_xor(s, off, 32);
        float mn   = fmaxf(m, s);
        float corr = __expf(m - mn);
        float p    = __expf(s - mn);
        l   = l   * corr + p;
        acc = acc * corr + p * Vb[base];
        m = mn;
    }
    O[((size_t)(b * Q_ + q)) * C_ + h * DH_ + lane] = acc / l;
}

// ---------------------------------------------------------------------------
// Deformable attention core: one wave per (b,q,h), lane = channel d.
// Fuses softmax over the 16 (level,point) weights + bilinear gather-sum.
// value layout: (B, LV, NH*DH) row-major (== reference pre-transpose view).
// ---------------------------------------------------------------------------
__device__ __forceinline__ float dfm_corner(const float* __restrict__ value,
                                            size_t bbase, int s0, int H_, int W_,
                                            int yi, int xi, int hch)
{
    bool valid = (yi >= 0) && (yi < H_) && (xi >= 0) && (xi < W_);
    int yc = imin(imax(yi, 0), H_ - 1);
    int xc = imin(imax(xi, 0), W_ - 1);
    float v = value[(bbase + (size_t)(s0 + yc * W_ + xc)) * C_ + hch];
    return valid ? v : 0.f;
}

__global__ void __launch_bounds__(256)
deform_attn_kernel(const float* __restrict__ offs, const float* __restrict__ awl,
                   const float* __restrict__ refp, const float* __restrict__ vrat,
                   const float* __restrict__ value, float* __restrict__ O)
{
    const int HW[4]  = {128, 64, 32, 16};
    const int LSI[4] = {0, 16384, 20480, 21504};

    const int gw   = (blockIdx.x * blockDim.x + threadIdx.x) >> 5;
    const int lane = threadIdx.x & 31;
    if (gw >= B_ * Q_ * NH_) return;
    const int b = gw / (Q_ * NH_);
    const int r = gw % (Q_ * NH_);
    const int q = r / NH_;
    const int h = r % NH_;

    // softmax over the 16 attention weights (lane-redundant, cheap)
    const float* awp = awl + ((size_t)(b * Q_ + q)) * (NH_ * NL_ * NP_) + h * (NL_ * NP_);
    float w[16], mx = -INFINITY;
    #pragma unroll
    for (int i = 0; i < 16; ++i) { w[i] = awp[i]; mx = fmaxf(mx, w[i]); }
    float sum = 0.f;
    #pragma unroll
    for (int i = 0; i < 16; ++i) { w[i] = __expf(w[i] - mx); sum += w[i]; }
    const float inv = 1.f / sum;

    const float* op = offs + ((size_t)(b * Q_ + q)) * C_ + h * (NL_ * NP_ * 2);
    const float rx = refp[(size_t)(b * Q_ + q) * 2 + 0];
    const float ry = refp[(size_t)(b * Q_ + q) * 2 + 1];
    const int   hch = h * DH_ + lane;
    const size_t bbase = (size_t)b * LV_;

    float acc = 0.f;
    #pragma unroll
    for (int l = 0; l < NL_; ++l) {
        const int H_ = HW[l], W_ = HW[l], s0 = LSI[l];
        const float vrx = vrat[(b * NL_ + l) * 2 + 0];
        const float vry = vrat[(b * NL_ + l) * 2 + 1];
        const float bx = rx * vrx, by = ry * vry;
        #pragma unroll
        for (int p = 0; p < NP_; ++p) {
            float ox = op[(l * NP_ + p) * 2 + 0];
            float oy = op[(l * NP_ + p) * 2 + 1];
            float x = (bx + ox / (float)W_) * W_ - 0.5f;
            float y = (by + oy / (float)H_) * H_ - 0.5f;
            float xf = floorf(x), yf = floorf(y);
            float fx = x - xf, fy = y - yf;
            int x0 = (int)xf, y0 = (int)yf;
            float v00 = dfm_corner(value, bbase, s0, H_, W_, y0,     x0,     hch);
            float v01 = dfm_corner(value, bbase, s0, H_, W_, y0,     x0 + 1, hch);
            float v10 = dfm_corner(value, bbase, s0, H_, W_, y0 + 1, x0,     hch);
            float v11 = dfm_corner(value, bbase, s0, H_, W_, y0 + 1, x0 + 1, hch);
            float samp = v00 * (1 - fy) * (1 - fx) + v01 * (1 - fy) * fx
                       + v10 * fy * (1 - fx)       + v11 * fy * fx;
            acc += (w[l * NP_ + p] * inv) * samp;
        }
    }
    O[((size_t)(b * Q_ + q)) * C_ + hch] = acc;
}

// ---------------------------------------------------------------------------
// Host orchestration
// ---------------------------------------------------------------------------
extern "C" void kernel_launch(void* const* d_in, const int* in_sizes, int n_in,
                              void* d_out, int out_size, void* d_ws, size_t ws_size,
                              hipStream_t stream)
{
    (void)in_sizes; (void)n_in; (void)out_size; (void)ws_size;
    const float* tgt  = (const float*)d_in[0];
    const float* qpos = (const float*)d_in[1];
    const float* refp = (const float*)d_in[2];
    const float* src  = (const float*)d_in[3];
    const float* vrat = (const float*)d_in[4];
    // d_in[5] spatial_shapes, d_in[6] level_start_index: compile-time constants

    auto P = [&](int layer, int idx) -> const float* {
        return (const float*)d_in[7 + layer * 26 + idx];
    };
    // per-layer flat param order:
    // 0 q.w 1 q.b 2 k.w 3 k.b 4 v.w 5 v.b 6 o.w 7 o.b 8 n2g 9 n2b
    // 10 off.w 11 off.b 12 aw.w 13 aw.b 14 val.w 15 val.b 16 out.w 17 out.b
    // 18 n1g 19 n1b 20 f1.w 21 f1.b 22 f2.w 23 f2.b 24 n3g 25 n3b

    float* wsp = (float*)d_ws;
    size_t off_ = 0;
    auto alloc = [&](size_t n) { float* p = wsp + off_; off_ += n; return p; };
    float* xbuf  = alloc((size_t)BQ_ * C_);
    float* addb  = alloc((size_t)BQ_ * C_);
    float* qb    = alloc((size_t)BQ_ * C_);
    float* kb    = alloc((size_t)BQ_ * C_);
    float* vb    = alloc((size_t)BQ_ * C_);
    float* t2a   = alloc((size_t)BQ_ * C_);
    float* t2    = alloc((size_t)BQ_ * C_);
    float* offb  = alloc((size_t)BQ_ * C_);
    float* awb   = alloc((size_t)BQ_ * (NH_ * NL_ * NP_));
    float* msout = alloc((size_t)BQ_ * C_);
    float* ffnh  = alloc((size_t)BQ_ * DFF_);
    float* value = alloc((size_t)B_ * LV_ * C_);

    hipMemcpyAsync(xbuf, tgt, sizeof(float) * BQ_ * C_,
                   hipMemcpyDeviceToDevice, stream);

    auto gemm = [&](const float* X, const float* Wm, const float* bias, float* Y,
                    int M, int N, int K, int relu) {
        dim3 grid((M / 16 + 3) / 4, N / 64);
        wmma_gemm_bias<<<grid, dim3(128), 0, stream>>>(X, Wm, bias, Y, M, N, K, relu);
    };

    const int nElem    = BQ_ * C_;
    const int addGrid  = (nElem + 255) / 256;
    const int waveGrid = (B_ * NH_ * Q_ * 32 + 255) / 256;

    for (int L = 0; L < NLAYERS_; ++L) {
        // ---- self attention ----
        add_kernel<<<addGrid, 256, 0, stream>>>(addb, xbuf, qpos, nElem);
        gemm(addb, P(L, 0), P(L, 1), qb, BQ_, C_, C_, 0);
        gemm(addb, P(L, 2), P(L, 3), kb, BQ_, C_, C_, 0);
        gemm(xbuf, P(L, 4), P(L, 5), vb, BQ_, C_, C_, 0);
        self_attn_kernel<<<waveGrid, 256, 0, stream>>>(qb, kb, vb, t2a);
        gemm(t2a, P(L, 6), P(L, 7), t2, BQ_, C_, C_, 0);
        res_ln_kernel<<<BQ_, 256, 0, stream>>>(xbuf, t2, P(L, 8), P(L, 9), xbuf);

        // ---- MS deformable attention ----
        add_kernel<<<addGrid, 256, 0, stream>>>(addb, xbuf, qpos, nElem);
        gemm(src,  P(L, 14), P(L, 15), value, B_ * LV_, C_, C_, 0);  // dominant GEMM
        gemm(addb, P(L, 10), P(L, 11), offb,  BQ_, C_, C_, 0);
        gemm(addb, P(L, 12), P(L, 13), awb,   BQ_, NH_ * NL_ * NP_, C_, 0);
        deform_attn_kernel<<<waveGrid, 256, 0, stream>>>(offb, awb, refp, vrat,
                                                         value, msout);
        gemm(msout, P(L, 16), P(L, 17), t2, BQ_, C_, C_, 0);
        res_ln_kernel<<<BQ_, 256, 0, stream>>>(xbuf, t2, P(L, 18), P(L, 19), xbuf);

        // ---- FFN ----
        gemm(xbuf, P(L, 20), P(L, 21), ffnh, BQ_, DFF_, C_, 1);
        gemm(ffnh, P(L, 22), P(L, 23), t2,   BQ_, C_, DFF_, 0);
        res_ln_kernel<<<BQ_, 256, 0, stream>>>(xbuf, t2, P(L, 24), P(L, 25), xbuf);
    }

    hipMemcpyAsync(d_out, xbuf, sizeof(float) * BQ_ * C_,
                   hipMemcpyDeviceToDevice, stream);
}
